// SolidGNN_29618094473954
// MI455X (gfx1250) — matrix operand order repaired and checked by reading, not compiled, not run on hardware
//
#include <hip/hip_runtime.h>
#include <hip/hip_bf16.h>

// ---------------------------------------------------------------------------
// GNN message passing for MI455X (gfx1250), bf16 WMMA path, f32 accumulate.
// h kept twice: f32 master (exact residual) + bf16 mirror (WMMA A operands).
// Weights pre-swizzled once into WMMA-B fragment order (bf16) in workspace;
// per-block staging via Tensor Data Mover (tensor_load_to_lds) if available.
// Edge kernel: 32-edge M-tile per wave -> each LDS B-fragment feeds 2 WMMAs.
// ---------------------------------------------------------------------------
#define GNN_B 2
#define GNN_N 20000
#define GNN_E 320000
#define GNN_D 64
#define GNN_L 25

typedef __attribute__((ext_vector_type(16))) __bf16 v16bf;
typedef __attribute__((ext_vector_type(8)))  __bf16 v8bf;
typedef __attribute__((ext_vector_type(8)))  float  v8f;

__device__ __forceinline__ __bf16 f2bf(float f) {
  unsigned u = __builtin_bit_cast(unsigned, f);
  unsigned r = u + 0x7fffu + ((u >> 16) & 1u);   // round-to-nearest-even
  unsigned short hs = (unsigned short)(r >> 16);
  return __builtin_bit_cast(__bf16, hs);
}

// A-matrix 16x32 bf16 (ISA 7.12.2): lane holds row (lane&15);
// element e -> K = 8*(lane>>4) + e + (e>=8 ? 8 : 0)  == two 8-elem runs.
__device__ __forceinline__ v16bf loadA_bf(const __bf16* base, int hf) {
  v8bf lo = *(const v8bf*)(base + 8 * hf);        // K = 8*hf .. +7
  v8bf hi = *(const v8bf*)(base + 8 * hf + 16);   // K = 8*hf+16 .. +23
  v16bf a;
#pragma unroll
  for (int e = 0; e < 8; ++e) { a[e] = lo[e]; a[e + 8] = hi[e]; }
  return a;
}

// WMMA-B fragment-swizzled layout: block fb = kc*4+nt, 32 lanes x 16 elems,
// lane stride 24 elems (48B, 16B aligned, conflict-padded).
// B layout: lane holds col (lane&15), element e -> K = 8*(lane>>4) + e + (e>=8 ? 8 : 0), as for A.
#define FRAG_LSTRIDE 24
#define FRAG_ELEMS   (32 * FRAG_LSTRIDE)
#define W1F_ELEMS (5 * 4 * FRAG_ELEMS)   // 15360
#define W2F_ELEMS (2 * 4 * FRAG_ELEMS)   //  6144
#define UWF_ELEMS (4 * 4 * FRAG_ELEMS)   // 12288
#define LAYER_W_ELEMS (W1F_ELEMS + W2F_ELEMS + UWF_ELEMS)   // 33792

__device__ __forceinline__ v16bf loadB_frag(const __bf16* frag, int kc, int nt,
                                            int lane) {
  return *(const v16bf*)(frag + (kc * 4 + nt) * FRAG_ELEMS +
                         lane * FRAG_LSTRIDE);   // 32B -> 2x ds_load_b128
}

// ---------------------------------------------------------------------------
// TDM copy: one 1D descriptor, issued by one wave. Falls back to vector copy.
// ---------------------------------------------------------------------------
#if __has_builtin(__builtin_amdgcn_tensor_load_to_lds)
#define GNN_HAVE_TDM 1
typedef unsigned int v4u_t __attribute__((ext_vector_type(4)));
typedef int v8i_t __attribute__((ext_vector_type(8)));
typedef int v4i_t __attribute__((ext_vector_type(4)));

__device__ __forceinline__ void tdm_copy_to_lds(void* lds, const void* gsrc,
                                                unsigned bytes) {
  const unsigned long long ga = (unsigned long long)gsrc;
  const unsigned ldsoff = (unsigned)(unsigned long long)lds;  // low 32 = LDS addr
  const unsigned n8 = bytes >> 3;                             // 8-byte elements
  // D# group 0 (ISA 8.3): count=1 | lds_addr | global_addr[56:0] | type=2
  v4u_t g0;
  g0[0] = 1u;
  g0[1] = ldsoff;
  g0[2] = (unsigned)ga;
  g0[3] = (unsigned)((ga >> 32) & 0x01ffffffu) | (2u << 30);
  // D# group 1 (ISA 8.4): data_size=3 (8B); tensor_dim0 = tile_dim0 = n8;
  // tensor_dim1 = tile_dim1 = 1; stride0 = n8.
  v8i_t g1;
  g1[0] = (int)(3u << 16);
  g1[1] = (int)((n8 & 0xffffu) << 16);
  g1[2] = (int)((n8 >> 16) | (1u << 16));
  g1[3] = (int)((n8 & 0xffffu) << 16);
  g1[4] = 1;
  g1[5] = (int)n8;
  g1[6] = 0;
  g1[7] = 0;
  v4i_t z4 = {0, 0, 0, 0};
#if defined(__clang_major__) && (__clang_major__ >= 23)
  v8i_t z8 = {0, 0, 0, 0, 0, 0, 0, 0};
  __builtin_amdgcn_tensor_load_to_lds(g0, g1, z4, z4, z8, 0);
#else
  __builtin_amdgcn_tensor_load_to_lds(g0, g1, z4, z4, 0);
#endif
}
#endif

__device__ __forceinline__ void stage_weights(void* lds, const void* gsrc,
                                              unsigned bytes, int tid) {
#ifdef GNN_HAVE_TDM
  if (tid < 32) {
    tdm_copy_to_lds(lds, gsrc, bytes);
    __builtin_amdgcn_s_wait_tensorcnt(0);
  }
  __syncthreads();
#else
  const uint4* s = (const uint4*)gsrc;
  uint4* d = (uint4*)lds;
  for (unsigned i = tid; i < (bytes >> 4); i += 256) d[i] = s[i];
  __syncthreads();
#endif
}

// ---------------------------------------------------------------------------
// Weight pre-swizzle: f32 [rows x 64] -> bf16 WMMA-B fragment order, once.
// ---------------------------------------------------------------------------
__device__ __forceinline__ __bf16 frag_elem(const float* w, int o,
                                            int validRows) {
  const int e = o % FRAG_LSTRIDE;
  if (e >= 16) return f2bf(0.0f);                  // lane pad
  const int lane = (o / FRAG_LSTRIDE) & 31;
  const int fb = o / FRAG_ELEMS;
  const int kc = fb >> 2, nt = fb & 3;
  const int row = kc * 32 + 8 * (lane >> 4) + e + ((e >= 8) ? 8 : 0);  // kmapB
  const float v = (row < validRows) ? w[row * 64 + nt * 16 + (lane & 15)]
                                    : 0.0f;
  return f2bf(v);
}

__global__ __launch_bounds__(256) void gnn_prep_weights(
    const float* __restrict__ msg_w1, const float* __restrict__ msg_w2,
    const float* __restrict__ up_w, __bf16* __restrict__ wsw) {
  long i = (long)blockIdx.x * 256 + threadIdx.x;
  const long tot = (long)GNN_L * LAYER_W_ELEMS;
  if (i >= tot) return;
  const int l = (int)(i / LAYER_W_ELEMS);
  const int o = (int)(i % LAYER_W_ELEMS);
  __bf16 v;
  if (o < W1F_ELEMS)
    v = frag_elem(msg_w1 + (size_t)l * 130 * 64, o, 130);
  else if (o < W1F_ELEMS + W2F_ELEMS)
    v = frag_elem(msg_w2 + (size_t)l * 64 * 64, o - W1F_ELEMS, 64);
  else
    v = frag_elem(up_w + (size_t)l * 128 * 64, o - W1F_ELEMS - W2F_ELEMS, 128);
  wsw[i] = v;
}

// ---------------------------------------------------------------------------
// Encoder: h = x @ enc_w + enc_b  (also writes bf16 mirror)
// ---------------------------------------------------------------------------
__global__ __launch_bounds__(256) void gnn_enc_kernel(
    const float* __restrict__ x, const float* __restrict__ ew,
    const float* __restrict__ eb, float* __restrict__ h,
    __bf16* __restrict__ hbf) {
  long i = (long)blockIdx.x * 256 + threadIdx.x;
  const long tot = (long)GNN_B * GNN_N * GNN_D;
  if (i >= tot) return;
  int dd = (int)(i & 63);
  long node = i >> 6;
  const float* xp = x + node * 3;
  float v = xp[0] * ew[dd] + xp[1] * ew[64 + dd] + xp[2] * ew[128 + dd] + eb[dd];
  h[i] = v;
  hbf[i] = f2bf(v);
}

// ---------------------------------------------------------------------------
// Edge kernel: per 32-edge tile (one wave, two 16-row WMMA sub-tiles):
//   cat = [h[row] | h[col] | edge_attr]  (K = 130, padded to 160)
//   m   = relu(LN(cat @ w1 + b1)) @ w2 + b2 ; atomicAdd into aggr[row]
// ---------------------------------------------------------------------------
#define STAGE_RS 72   // stage row stride in bf16 elems (144B, conflict pad)
__global__ __launch_bounds__(256) void gnn_edge_kernel(
    const __bf16* __restrict__ hbf, float* __restrict__ aggr,
    const int* __restrict__ eidx, const float* __restrict__ eattr,
    const __bf16* __restrict__ wfsrc,   // pre-swizzled w1f|w2f for this layer
    const float* __restrict__ b1, const float* __restrict__ g1,
    const float* __restrict__ be1, const float* __restrict__ b2) {
  __shared__ __align__(16) __bf16 wfLDS[W1F_ELEMS + W2F_ELEMS];  // 42 KB
  __shared__ __align__(16) __bf16 stage[8][16 * STAGE_RS];       // 18 KB

  const int tid = threadIdx.x;
  stage_weights(wfLDS, wfsrc, (W1F_ELEMS + W2F_ELEMS) * 2, tid);
  const __bf16* w1f = wfLDS;
  const __bf16* w2f = wfLDS + W1F_ELEMS;

  const int wave = tid >> 5, lane = tid & 31;
  const int hf = lane >> 4, nlo = lane & 15;
  const int g = blockIdx.y;
  const __bf16* hg = hbf + (size_t)g * GNN_N * GNN_D;
  float* ag = aggr + (size_t)g * GNN_N * GNN_D;
  __bf16* stg = stage[wave];

  const int tileId = blockIdx.x * 8 + wave;   // E/32 tiles, exact (no tail)
  const int e0 = tileId * 32;

  int myRow[2], myCol[2];
  float at0[2], at1[2];
#pragma unroll
  for (int t = 0; t < 2; ++t) {
    const int em = e0 + t * 16 + nlo;         // this lane's A-row edge
    myRow[t] = eidx[em];
    myCol[t] = eidx[GNN_E + em];
    at0[t] = eattr[(size_t)em * 2 + 0];
    at1[t] = eattr[(size_t)em * 2 + 1];
  }

  // ---- GEMM 1: cat @ w1; each B fragment feeds both sub-tiles ----
  v8f acc[2][4] = {{v8f{}, v8f{}, v8f{}, v8f{}}, {v8f{}, v8f{}, v8f{}, v8f{}}};
#pragma unroll
  for (int kc = 0; kc < 5; ++kc) {
    v16bf a[2];
#pragma unroll
    for (int t = 0; t < 2; ++t) {
      if (kc < 4) {
        const __bf16* src =
            hg + (size_t)((kc < 2) ? myRow[t] : myCol[t]) * GNN_D +
            (kc & 1) * 32;
        a[t] = loadA_bf(src, hf);              // 2x global_load_b128
      } else {
        union { v16bf v; unsigned u[8]; } au;
#pragma unroll
        for (int w = 0; w < 8; ++w) au.u[w] = 0u;
        if (hf == 0) {                         // K=128,129 -> e=0,1, half 0
          unsigned p =
              (unsigned)__builtin_bit_cast(unsigned short, f2bf(at0[t])) |
              ((unsigned)__builtin_bit_cast(unsigned short, f2bf(at1[t])) << 16);
          au.u[0] = p;
        }
        a[t] = au.v;
      }
    }
#pragma unroll
    for (int nt = 0; nt < 4; ++nt) {
      v16bf bm = loadB_frag(w1f, kc, nt, lane);   // 2x ds_load_b128
      acc[0][nt] = __builtin_amdgcn_wmma_f32_16x16x32_bf16(
          false, a[0], false, bm, (short)0, acc[0][nt], false, false);
      acc[1][nt] = __builtin_amdgcn_wmma_f32_16x16x32_bf16(
          false, a[1], false, bm, (short)0, acc[1][nt], false, false);
    }
  }

  // per-lane output-column constants
  float bv[4], gv[4], bev[4], b2v[4];
#pragma unroll
  for (int nt = 0; nt < 4; ++nt) {
    bv[nt] = b1[nt * 16 + nlo];
    gv[nt] = g1[nt * 16 + nlo];
    bev[nt] = be1[nt * 16 + nlo];
    b2v[nt] = b2[nt * 16 + nlo];
  }

  // ---- per sub-tile: bias + LN + relu -> stage -> GEMM2 -> scatter ----
#pragma unroll
  for (int t = 0; t < 2; ++t) {
    float sum[8], sq[8];
#pragma unroll
    for (int r = 0; r < 8; ++r) { sum[r] = 0.f; sq[r] = 0.f; }
#pragma unroll
    for (int nt = 0; nt < 4; ++nt) {
#pragma unroll
      for (int r = 0; r < 8; ++r) {
        float v = acc[t][nt][r] + bv[nt];
        acc[t][nt][r] = v;
        sum[r] += v; sq[r] += v * v;
      }
    }
#pragma unroll
    for (int mask = 1; mask <= 8; mask <<= 1) {   // reduce in 16-lane half
#pragma unroll
      for (int r = 0; r < 8; ++r) {
        sum[r] += __shfl_xor(sum[r], mask, 32);
        sq[r]  += __shfl_xor(sq[r],  mask, 32);
      }
    }
    float mean[8], rstd[8];
#pragma unroll
    for (int r = 0; r < 8; ++r) {
      mean[r] = sum[r] * (1.0f / 64.0f);
      float var = sq[r] * (1.0f / 64.0f) - mean[r] * mean[r];
      rstd[r] = rsqrtf(var + 1e-5f);
    }

    // normalize + affine + relu; C-layout -> row-major reshape via LDS
    asm volatile("" ::: "memory");
#pragma unroll
    for (int nt = 0; nt < 4; ++nt) {
#pragma unroll
      for (int r = 0; r < 8; ++r) {
        float y = (acc[t][nt][r] - mean[r]) * rstd[r] * gv[nt] + bev[nt];
        y = y > 0.f ? y : 0.f;
        stg[(r + 8 * hf) * STAGE_RS + nt * 16 + nlo] = f2bf(y);
      }
    }
    asm volatile("s_wait_dscnt 0" ::: "memory");  // intra-wave LDS RAW

    // GEMM 2: m @ w2 (K = 64)
    v8f dacc[4] = {v8f{}, v8f{}, v8f{}, v8f{}};
#pragma unroll
    for (int kc = 0; kc < 2; ++kc) {
      v16bf am = loadA_bf(stg + nlo * STAGE_RS + kc * 32, hf);
#pragma unroll
      for (int nt = 0; nt < 4; ++nt) {
        v16bf bm = loadB_frag(w2f, kc, nt, lane);
        dacc[nt] = __builtin_amdgcn_wmma_f32_16x16x32_bf16(
            false, am, false, bm, (short)0, dacc[nt], false, false);
      }
    }

    // scatter-add: aggr[row[e]] += m[e]  (f32 atomics, stays in L2)
    int rows[8];
#pragma unroll
    for (int r = 0; r < 8; ++r) rows[r] = eidx[e0 + t * 16 + r + 8 * hf];
#pragma unroll
    for (int nt = 0; nt < 4; ++nt) {
      const int n = nt * 16 + nlo;
#pragma unroll
      for (int r = 0; r < 8; ++r)
        atomicAdd(&ag[(size_t)rows[r] * GNN_D + n], dacc[nt][r] + b2v[nt]);
    }
    asm volatile("" ::: "memory");   // keep tile-1 stage stores after reads
  }
}

// ---------------------------------------------------------------------------
// Update kernel: u = relu(LN([h | aggr] @ uw + ub)); h += u
// (writes both f32 master and bf16 mirror of h)
// ---------------------------------------------------------------------------
__global__ __launch_bounds__(256) void gnn_update_kernel(
    float* __restrict__ h, __bf16* __restrict__ hbf,
    const float* __restrict__ aggr,
    const __bf16* __restrict__ uwfsrc,   // pre-swizzled uw for this layer
    const float* __restrict__ ub, const float* __restrict__ ug,
    const float* __restrict__ ube) {
  __shared__ __align__(16) __bf16 uwf[UWF_ELEMS];   // 24 KB
  const int tid = threadIdx.x;
  stage_weights(uwf, uwfsrc, UWF_ELEMS * 2, tid);

  const int wave = tid >> 5, lane = tid & 31;
  const int hf = lane >> 4, nlo = lane & 15;
  const int g = blockIdx.y;
  float* hg = h + (size_t)g * GNN_N * GNN_D;
  __bf16* hbg = hbf + (size_t)g * GNN_N * GNN_D;
  const float* ag = aggr + (size_t)g * GNN_N * GNN_D;

  const int tileId = blockIdx.x * 8 + wave;
  const int n0 = tileId * 16;
  int node = n0 + nlo;
  if (node >= GNN_N) node = GNN_N - 1;   // clamp for tail tiles (reads only)

  v8f acc[4] = {v8f{}, v8f{}, v8f{}, v8f{}};
#pragma unroll
  for (int kc = 0; kc < 4; ++kc) {
    v16bf a;
    if (kc < 2) {
      a = loadA_bf(hbg + (size_t)node * GNN_D + kc * 32, hf);
    } else {
      const float* src = ag + (size_t)node * GNN_D + (kc - 2) * 32;
#pragma unroll
      for (int e = 0; e < 16; ++e)
        a[e] = f2bf(src[8 * hf + e + ((e >= 8) ? 8 : 0)]);
    }
#pragma unroll
    for (int nt = 0; nt < 4; ++nt) {
      v16bf bm = loadB_frag(uwf, kc, nt, lane);
      acc[nt] = __builtin_amdgcn_wmma_f32_16x16x32_bf16(
          false, a, false, bm, (short)0, acc[nt], false, false);
    }
  }

  float sum[8], sq[8];
#pragma unroll
  for (int r = 0; r < 8; ++r) { sum[r] = 0.f; sq[r] = 0.f; }
#pragma unroll
  for (int nt = 0; nt < 4; ++nt) {
    const float bb = ub[nt * 16 + nlo];
#pragma unroll
    for (int r = 0; r < 8; ++r) {
      float v = acc[nt][r] + bb;
      acc[nt][r] = v;
      sum[r] += v; sq[r] += v * v;
    }
  }
#pragma unroll
  for (int mask = 1; mask <= 8; mask <<= 1) {
#pragma unroll
    for (int r = 0; r < 8; ++r) {
      sum[r] += __shfl_xor(sum[r], mask, 32);
      sq[r]  += __shfl_xor(sq[r],  mask, 32);
    }
  }
  float mean[8], rstd[8];
#pragma unroll
  for (int r = 0; r < 8; ++r) {
    mean[r] = sum[r] * (1.0f / 64.0f);
    float var = sq[r] * (1.0f / 64.0f) - mean[r] * mean[r];
    rstd[r] = rsqrtf(var + 1e-5f);
  }

  // residual: h[node] += relu(LN(...)); refresh bf16 mirror
#pragma unroll
  for (int nt = 0; nt < 4; ++nt) {
    const int n = nt * 16 + nlo;
    const float gg = ug[n];
    const float bb = ube[n];
#pragma unroll
    for (int r = 0; r < 8; ++r) {
      const int nd = n0 + r + 8 * hf;
      if (nd < GNN_N) {
        float y = (acc[nt][r] - mean[r]) * rstd[r] * gg + bb;
        y = y > 0.f ? y : 0.f;
        const size_t off = (size_t)nd * GNN_D + n;
        const float hv = hg[off] + y;
        hg[off] = hv;
        hbg[off] = f2bf(hv);
      }
    }
  }
}

// ---------------------------------------------------------------------------
// Decoder: out = relu(h @ dec_w1 + b1) @ dec_w2 + b2
// ---------------------------------------------------------------------------
__global__ __launch_bounds__(256) void gnn_dec_kernel(
    const float* __restrict__ h, const float* __restrict__ w1,
    const float* __restrict__ b1, const float* __restrict__ w2,
    const float* __restrict__ b2, float* __restrict__ out) {
  long node = (long)blockIdx.x * 256 + threadIdx.x;
  if (node >= (long)GNN_B * GNN_N) return;
  const float* hp = h + node * GNN_D;
  float o0 = b2[0], o1 = b2[1];
  for (int j = 0; j < 32; ++j) {
    float t = b1[j];
#pragma unroll 8
    for (int k = 0; k < GNN_D; ++k) t += hp[k] * w1[k * 32 + j];
    t = t > 0.f ? t : 0.f;
    o0 += t * w2[j * 2 + 0];
    o1 += t * w2[j * 2 + 1];
  }
  out[node * 2 + 0] = o0;
  out[node * 2 + 1] = o1;
}

// ---------------------------------------------------------------------------
extern "C" void kernel_launch(void* const* d_in, const int* in_sizes, int n_in,
                              void* d_out, int out_size, void* d_ws,
                              size_t ws_size, hipStream_t stream) {
  (void)in_sizes; (void)n_in; (void)out_size; (void)ws_size;
  const float* x      = (const float*)d_in[0];
  const int*   eidx   = (const int*)  d_in[1];
  const float* eattr  = (const float*)d_in[2];
  const float* enc_w  = (const float*)d_in[3];
  const float* enc_b  = (const float*)d_in[4];
  const float* msg_w1 = (const float*)d_in[5];
  const float* msg_b1 = (const float*)d_in[6];
  const float* msg_g  = (const float*)d_in[7];
  const float* msg_be = (const float*)d_in[8];
  const float* msg_w2 = (const float*)d_in[9];
  const float* msg_b2 = (const float*)d_in[10];
  const float* up_w   = (const float*)d_in[11];
  const float* up_b   = (const float*)d_in[12];
  const float* up_g   = (const float*)d_in[13];
  const float* up_be  = (const float*)d_in[14];
  const float* dec_w1 = (const float*)d_in[15];
  const float* dec_b1 = (const float*)d_in[16];
  const float* dec_w2 = (const float*)d_in[17];
  const float* dec_b2 = (const float*)d_in[18];
  float* out = (float*)d_out;

  const size_t hElems = (size_t)GNN_B * GNN_N * GNN_D;   // 2.56M
  float*  h    = (float*)d_ws;                 // [B][N][64] f32 master
  float*  aggr = h + hElems;                   // [B][N][64] f32
  __bf16* hbf  = (__bf16*)(aggr + hElems);     // [B][N][64] bf16 mirror
  __bf16* wsw  = hbf + hElems;                 // [L][33792] swizzled weights

  {
    const long tot = (long)GNN_L * LAYER_W_ELEMS;
    gnn_prep_weights<<<(int)((tot + 255) / 256), 256, 0, stream>>>(
        msg_w1, msg_w2, up_w, wsw);
  }
  {
    const long tot = (long)GNN_B * GNN_N * GNN_D;
    gnn_enc_kernel<<<(int)((tot + 255) / 256), 256, 0, stream>>>(
        x, enc_w, enc_b, h, hbf);
  }

  for (int l = 0; l < GNN_L; ++l) {
    const __bf16* wl = wsw + (size_t)l * LAYER_W_ELEMS;
    hipMemsetAsync(aggr, 0, hElems * sizeof(float), stream);
    gnn_edge_kernel<<<dim3(GNN_E / 256, GNN_B), 256, 0, stream>>>(
        hbf, aggr, eidx, eattr, wl,
        msg_b1 + (size_t)l * 64, msg_g + (size_t)l * 64,
        msg_be + (size_t)l * 64, msg_b2 + (size_t)l * 64);
    gnn_update_kernel<<<dim3((GNN_N / 16 + 7) / 8, GNN_B), 256, 0, stream>>>(
        h, hbf, aggr, wl + W1F_ELEMS + W2F_ELEMS,
        up_b + (size_t)l * 64, up_g + (size_t)l * 64,
        up_be + (size_t)l * 64);
  }

  gnn_dec_kernel<<<(GNN_B * GNN_N + 255) / 256, 256, 0, stream>>>(
      h, dec_w1, dec_b1, dec_w2, dec_b2, out);
}
